// BiDAF_80650895884742
// MI455X (gfx1250) — matrix operand order repaired and checked by reading, not compiled, not run on hardware
//
#include <hip/hip_runtime.h>
#include <hip/hip_bf16.h>
#include <math.h>

typedef __attribute__((ext_vector_type(16))) _Float16 v16h;
typedef __attribute__((ext_vector_type(8)))  float    v8f;

#define BS      32
#define MCL     1024
#define MQL     64
#define DDIM    512
#define CTILE   128     // 8 waves * 16 rows
#define DCHUNK  128
#define NEGINF  (-1e30f)

// ---------------------------------------------------------------------------
// Kernel 1: per (batch, context-tile of 128 rows).
//   s = (Hc*Wm) @ Hq^T + cdot + qdot + b  (WMMA f16, f32 accum)
//   softmax over q, row-max -> score workspace
//   caq = w @ Hq                          (WMMA f16, f32 accum)
//   out[:, 0:512]    = tanh(Hc)
//   out[:, 512:1024] = tanh(caq)
//   out[:,1024:1536] = tanh(Hc*caq)
// ---------------------------------------------------------------------------
__launch_bounds__(256)
__global__ void bidaf_attn_kernel(const float* __restrict__ Hq,
                                  const float* __restrict__ Hc,
                                  const unsigned char* __restrict__ qmask,
                                  const unsigned char* __restrict__ cmask,
                                  const float* __restrict__ W,
                                  const float* __restrict__ bias_p,
                                  float* __restrict__ out,
                                  float* __restrict__ score)
{
    __shared__ float sWm[DDIM];                         // 2 KB
    __shared__ float sCdot[CTILE];                      // 0.5 KB
    __shared__ float sQdot[MQL];                        // 0.25 KB
    __shared__ _Float16 sW[8][16][MQL];                 // 16 KB softmax weights
    __shared__ __align__(32) _Float16 sHqT[DCHUNK][MQL];// 16 KB Hq^T chunk (f16)

    const int tid  = threadIdx.x;
    const int lane = tid & 31;
    const int wave = tid >> 5;
    const int hi   = lane >> 4;        // which 16-lane half
    const int lq   = lane & 15;        // lane within half
    const int b    = blockIdx.y;
    const int cbase = blockIdx.x * CTILE;
    const int crow0 = cbase + wave * 16;   // first context row owned by this wave

    // ---- stage Wm, cdot, qdot --------------------------------------------
    for (int i = tid; i < DDIM; i += 256) sWm[i] = W[2 * DDIM + i];
    __syncthreads();

    for (int r = tid; r < CTILE; r += 256) {
        const float* row = Hc + ((size_t)b * MCL + cbase + r) * DDIM;
        float acc = 0.f;
        for (int d = 0; d < DDIM; ++d) acc += row[d] * W[d];        // Wc
        sCdot[r] = acc;
    }
    for (int q = tid; q < MQL; q += 256) {
        const float* row = Hq + ((size_t)b * MQL + q) * DDIM;
        float acc = 0.f;
        for (int d = 0; d < DDIM; ++d) acc += row[d] * W[DDIM + d]; // Wq
        sQdot[q] = acc;
    }
    __syncthreads();

    // ---- GEMM 1: s[16c x 64q] per wave, K = 512 --------------------------
    v8f acc[4] = {v8f{}, v8f{}, v8f{}, v8f{}};
    const float* hcrow = Hc + ((size_t)b * MCL + crow0 + lq) * DDIM;

    for (int kt = 0; kt < 16; ++kt) {
        // A fragment (16x32 f16): lane = row lq, two contiguous 8-elem runs
        const int d0 = kt * 32 + 16 * hi;
        v16h a;
        #pragma unroll
        for (int j = 0; j < 8; ++j) a[j]     = (_Float16)(hcrow[d0 + j]      * sWm[d0 + j]);
        #pragma unroll
        for (int j = 0; j < 8; ++j) a[8 + j] = (_Float16)(hcrow[d0 + 8 + j] * sWm[d0 + 8 + j]);

        // B fragments (32x16 f16): column q = nt*16+lq, 16 contiguous d
        #pragma unroll
        for (int nt = 0; nt < 4; ++nt) {
            const int q  = nt * 16 + lq;
            const int ds = kt * 32 + 16 * hi;
            const float* hqrow = Hq + ((size_t)b * MQL + q) * DDIM + ds;
            v16h bb;
            #pragma unroll
            for (int j = 0; j < 16; ++j) bb[j] = (_Float16)hqrow[j];
            acc[nt] = __builtin_amdgcn_wmma_f32_16x16x32_f16(
                false, a, false, bb, (short)0, acc[nt], false, false);
        }
    }

    // ---- bias + rank-1 terms + mask + row softmax over q -----------------
    const float bval = bias_p[0];
    float qadd[4]; bool qm[4];
    #pragma unroll
    for (int nt = 0; nt < 4; ++nt) {
        const int q = nt * 16 + lq;
        qadd[nt] = sQdot[q];
        qm[nt]   = qmask[b * MQL + q] != 0;
    }

    #pragma unroll
    for (int v = 0; v < 8; ++v) {
        const int m   = v + 8 * hi;                 // row within wave tile
        const int row = crow0 + m;
        const float cadd = sCdot[wave * 16 + m];
        const bool  cm   = cmask[b * MCL + row] != 0;

        float vals[4];
        #pragma unroll
        for (int nt = 0; nt < 4; ++nt) {
            float sv = acc[nt][v] + cadd + qadd[nt] + bval;
            if (!(cm && qm[nt])) sv = NEGINF;
            vals[nt] = sv;
        }
        float mx = fmaxf(fmaxf(vals[0], vals[1]), fmaxf(vals[2], vals[3]));
        #pragma unroll
        for (int off = 1; off < 16; off <<= 1)
            mx = fmaxf(mx, __shfl_xor(mx, off, 16));

        float se = 0.f;
        #pragma unroll
        for (int nt = 0; nt < 4; ++nt) { vals[nt] = expf(vals[nt] - mx); se += vals[nt]; }
        #pragma unroll
        for (int off = 1; off < 16; off <<= 1)
            se += __shfl_xor(se, off, 16);

        if (lq == 0) score[(size_t)b * MCL + row] = mx;  // score_q2c = row max

        const float inv = 1.0f / se;
        #pragma unroll
        for (int nt = 0; nt < 4; ++nt)
            sW[wave][m][nt * 16 + lq] = (_Float16)(vals[nt] * inv);
    }
    __syncthreads();

    // ---- load w back as A fragments (16x64, K = q) -----------------------
    v16h a2[2];
    #pragma unroll
    for (int kt2 = 0; kt2 < 2; ++kt2) {
        v16h t;
        #pragma unroll
        for (int h = 0; h < 16; ++h) {
            const int q = kt2 * 32 + h + 16 * hi;
            t[h] = sW[wave][lq][q];
        }
        a2[kt2] = t;
    }

    // ---- GEMM 2: caq = w @ Hq, chunked over d; fused tanh output ---------
    for (int ch = 0; ch < 4; ++ch) {
        const int dbase = ch * DCHUNK;
        __syncthreads();
        for (int e = tid; e < DCHUNK * MQL; e += 256) {
            const int dd = e % DCHUNK, q = e / DCHUNK;
            sHqT[dd][q] = (_Float16)Hq[((size_t)b * MQL + q) * DDIM + dbase + dd];
        }
        __syncthreads();

        #pragma unroll
        for (int nt2 = 0; nt2 < 8; ++nt2) {
            v8f c2 = v8f{};
            #pragma unroll
            for (int kt2 = 0; kt2 < 2; ++kt2) {
                const int qs = kt2 * 32 + 16 * hi;
                const int dd = nt2 * 16 + lq;
                v16h bb = *(const v16h*)&sHqT[dd][qs];   // 32B contiguous LDS
                c2 = __builtin_amdgcn_wmma_f32_16x16x32_f16(
                    false, a2[kt2], false, bb, (short)0, c2, false, false);
            }
            const int d = dbase + nt2 * 16 + lq;
            #pragma unroll
            for (int v = 0; v < 8; ++v) {
                const int row = crow0 + v + 8 * hi;
                const float caq = c2[v];
                const float hc  = Hc[((size_t)b * MCL + row) * DDIM + d];
                const size_t ob = ((size_t)b * MCL + row) * (size_t)(4 * DDIM);
                out[ob + d]            = tanhf(hc);
                out[ob + DDIM + d]     = tanhf(caq);
                out[ob + 2 * DDIM + d] = tanhf(hc * caq);
            }
        }
    }
}

// ---------------------------------------------------------------------------
// Kernel 2: per batch — softmax over score_q2c (c axis), qac = w @ Hc,
//           out[:,1536:2048] = tanh(Hc * qac)
// ---------------------------------------------------------------------------
__launch_bounds__(256)
__global__ void bidaf_q2c_kernel(const float* __restrict__ Hc,
                                 const float* __restrict__ score,
                                 float* __restrict__ out)
{
    __shared__ float sw[MCL];     // unnormalized weights, 4 KB
    __shared__ float sqac[DDIM];  // 2 KB
    __shared__ float red[8];
    __shared__ float bcast;

    const int b = blockIdx.x, tid = threadIdx.x;
    const int lane = tid & 31, wave = tid >> 5;

    // block max of score
    float mx = NEGINF;
    for (int i = tid; i < MCL; i += 256) mx = fmaxf(mx, score[(size_t)b * MCL + i]);
    #pragma unroll
    for (int off = 16; off > 0; off >>= 1) mx = fmaxf(mx, __shfl_xor(mx, off, 32));
    if (lane == 0) red[wave] = mx;
    __syncthreads();
    if (wave == 0) {
        float x = (lane < 8) ? red[lane] : NEGINF;
        #pragma unroll
        for (int off = 4; off > 0; off >>= 1) x = fmaxf(x, __shfl_xor(x, off, 32));
        if (lane == 0) bcast = x;
    }
    __syncthreads();
    const float M = bcast;
    __syncthreads();

    // exp + block sum
    float se = 0.f;
    for (int i = tid; i < MCL; i += 256) {
        float e = expf(score[(size_t)b * MCL + i] - M);
        sw[i] = e;
        se += e;
    }
    #pragma unroll
    for (int off = 16; off > 0; off >>= 1) se += __shfl_xor(se, off, 32);
    if (lane == 0) red[wave] = se;
    __syncthreads();
    if (wave == 0) {
        float x = (lane < 8) ? red[lane] : 0.f;
        #pragma unroll
        for (int off = 4; off > 0; off >>= 1) x += __shfl_xor(x, off, 32);
        if (lane == 0) bcast = x;
    }
    __syncthreads();
    const float invS = 1.0f / bcast;

    // qac[d] = sum_c w[c] * Hc[c][d]   (coalesced across threads)
    for (int d0 = tid; d0 < DDIM; d0 += 256) {
        float acc = 0.f;
        for (int c = 0; c < MCL; ++c)
            acc += sw[c] * Hc[((size_t)b * MCL + c) * DDIM + d0];
        sqac[d0] = acc * invS;
    }
    __syncthreads();

    // out[:,1536:2048] = tanh(Hc * qac)
    for (int e = tid; e < MCL * DDIM; e += 256) {
        const int c = e / DDIM, d = e % DDIM;
        const float hc = Hc[((size_t)b * MCL + c) * DDIM + d];
        out[((size_t)b * MCL + c) * (size_t)(4 * DDIM) + 3 * DDIM + d] = tanhf(hc * sqac[d]);
    }
}

// ---------------------------------------------------------------------------
extern "C" void kernel_launch(void* const* d_in, const int* in_sizes, int n_in,
                              void* d_out, int out_size, void* d_ws, size_t ws_size,
                              hipStream_t stream)
{
    (void)in_sizes; (void)n_in; (void)out_size; (void)ws_size;
    const float* Hq = (const float*)d_in[0];
    const float* Hc = (const float*)d_in[1];
    const unsigned char* qmask = (const unsigned char*)d_in[2];
    const unsigned char* cmask = (const unsigned char*)d_in[3];
    const float* W = (const float*)d_in[4];
    const float* bb = (const float*)d_in[5];
    float* out   = (float*)d_out;
    float* score = (float*)d_ws;   // 32*1024 floats

    dim3 g1(MCL / CTILE, BS);      // (8, 32)
    bidaf_attn_kernel<<<g1, 256, 0, stream>>>(Hq, Hc, qmask, cmask, W, bb, out, score);
    bidaf_q2c_kernel<<<BS, 256, 0, stream>>>(Hc, score, out);
}
